// MSDeformAttn_26018911879850
// MI455X (gfx1250) — matrix operand order repaired and checked by reading, not compiled, not run on hardware
//
#include <hip/hip_runtime.h>
#include <hip/hip_bf16.h>

// ---------------------------------------------------------------------------
// MSDeformAttn (3D) for MI455X / gfx1250, wave32.
// Big GEMMs on v_wmma_f32_16x16x32_f16 (f32 accumulate). Operands pre-packed
// into the CDNA5 WMMA register layouts so the GEMM does only contiguous
// 32B-per-lane fragment loads (served from the 192MB L2).
// ---------------------------------------------------------------------------

typedef __attribute__((ext_vector_type(16))) _Float16 v16h;
typedef __attribute__((ext_vector_type(8)))  float    v8f;

#define N_BATCH 8
#define LQ      4096
#define CDIM    768
#define HEADS   8
#define PNTS    4
#define DHEAD   96
#define NVOX    12
#define LIN     (NVOX * NVOX * NVOX)      // 1728
#define ROWS_Q  (N_BATCH * LQ)            // 32768
#define ROWS_V  (N_BATCH * LIN)           // 13824
#define KTILES  (CDIM / 32)               // 24

// ---------------------------------------------------------------------------
// Pack a row-major f32 matrix X[R][K] into f16 A-fragments.
// Tile (mt,kt) = 16x32 block -> 512 f16 laid out so lane L reads halfs
// [L*16 .. L*16+15] contiguously.  CDNA5 16-bit A layout (ISA 7.12.2):
//   lanes 0-15 : M=lane, v0-3 = K 0..7,  v4-7 = K 16..23
//   lanes16-31 : M=lane-16, v0-3 = K 8..15, v4-7 = K 24..31
// Tile order: mt-major, kt inner  (index = mt*KTILES + kt).
// ---------------------------------------------------------------------------
__global__ void pack_a_f16(const float* __restrict__ X, _Float16* __restrict__ P,
                           int Kcols, size_t total) {
  size_t tid = (size_t)blockIdx.x * blockDim.x + threadIdx.x;
  if (tid >= total) return;
  int    h    = (int)(tid & 15);
  int    lane = (int)((tid >> 4) & 31);
  size_t tile = tid >> 9;
  int    kt   = (int)(tile % KTILES);
  size_t mt   = tile / KTILES;
  int m   = lane & 15;
  int sub = lane >> 4;                 // 0: K-lo half, 1: K-hi half of 16
  int j   = h >> 1, par = h & 1;
  int k   = (j < 4) ? (sub * 8 + 2 * j + par)
                    : (16 + sub * 8 + 2 * (j - 4) + par);
  size_t row = mt * 16 + m;
  int    col = kt * 32 + k;
  P[tid] = (_Float16)X[row * (size_t)Kcols + col];
}

// ---------------------------------------------------------------------------
// Pack weights W[K][Nout] (row-major f32) into f16 B-fragments.
// Tile (kt,nt) = 32x16 block.  CDNA5 16-bit B layout:
//   lane n (0-15): column N=n, v0-3 = K 0..7,  v4-7 = K 16..23
//   lane n (16-31): column N=n-16, v0-3 = K 8..15, v4-7 = K 24..31
// Tile order: nt-major, kt inner  (index = nt*KTILES + kt).
// ---------------------------------------------------------------------------
__global__ void pack_b_f16(const float* __restrict__ W, _Float16* __restrict__ P,
                           int Ncols, size_t total) {
  size_t tid = (size_t)blockIdx.x * blockDim.x + threadIdx.x;
  if (tid >= total) return;
  int    h    = (int)(tid & 15);
  int    lane = (int)((tid >> 4) & 31);
  size_t tile = tid >> 9;
  int    kt   = (int)(tile % KTILES);
  size_t nt   = tile / KTILES;
  int n = lane & 15;
  int k = ((lane >> 4) ? 8 : 0) + ((h < 8) ? h : (16 + h - 8));
  size_t row = (size_t)(kt * 32 + k);
  size_t col = nt * 16 + n;
  P[tid] = (_Float16)W[row * (size_t)Ncols + col];
}

// ---------------------------------------------------------------------------
// C[M][Ncols] = packedA @ packedB + bias.  One wave -> 2x2 WMMA tiles (32x32).
// Workgroup = 8 waves arranged wavesM x wavesN (wavesM*wavesN == 8).
// All tile counts divide the grid exactly for our launches (no tail logic).
// ---------------------------------------------------------------------------
__global__ void wmma_gemm_f16(const _Float16* __restrict__ pA,
                              const _Float16* __restrict__ pB,
                              const float* __restrict__ bias,
                              float* __restrict__ C,
                              int Ncols, int wavesN) {
  const int wave = threadIdx.x >> 5;
  const int lane = threadIdx.x & 31;
  const int wm = wave / wavesN;
  const int wn = wave - wm * wavesN;
  const int wavesM = 8 / wavesN;
  const int mt0 = (blockIdx.y * wavesM + wm) * 2;
  const int nt0 = (blockIdx.x * wavesN + wn) * 2;

  const _Float16* a0p = pA + ((size_t)(mt0    ) * KTILES) * 512 + lane * 16;
  const _Float16* a1p = pA + ((size_t)(mt0 + 1) * KTILES) * 512 + lane * 16;
  const _Float16* b0p = pB + ((size_t)(nt0    ) * KTILES) * 512 + lane * 16;
  const _Float16* b1p = pB + ((size_t)(nt0 + 1) * KTILES) * 512 + lane * 16;

  v8f c00 = {}, c01 = {}, c10 = {}, c11 = {};
#pragma unroll 4
  for (int kt = 0; kt < KTILES; ++kt) {
    v16h a0 = *(const v16h*)(a0p + (size_t)kt * 512);
    v16h a1 = *(const v16h*)(a1p + (size_t)kt * 512);
    v16h b0 = *(const v16h*)(b0p + (size_t)kt * 512);
    v16h b1 = *(const v16h*)(b1p + (size_t)kt * 512);
    c00 = __builtin_amdgcn_wmma_f32_16x16x32_f16(false, a0, false, b0, (short)0, c00, false, false);
    c01 = __builtin_amdgcn_wmma_f32_16x16x32_f16(false, a0, false, b1, (short)0, c01, false, false);
    c10 = __builtin_amdgcn_wmma_f32_16x16x32_f16(false, a1, false, b0, (short)0, c10, false, false);
    c11 = __builtin_amdgcn_wmma_f32_16x16x32_f16(false, a1, false, b1, (short)0, c11, false, false);
  }

  // C/D layout: VGPR r holds (M = r + (lane<16?0:8), N = lane&15) of the tile.
  const int nloc = lane & 15;
  const int mofs = (lane & 16) ? 8 : 0;
  {
    const int col0 = nt0 * 16 + nloc, col1 = (nt0 + 1) * 16 + nloc;
    const float bv0 = bias[col0], bv1 = bias[col1];
    float* r0 = C + ((size_t)(mt0 * 16 + mofs)) * Ncols;
    float* r1 = C + ((size_t)((mt0 + 1) * 16 + mofs)) * Ncols;
#pragma unroll
    for (int r = 0; r < 8; ++r) {
      r0[(size_t)r * Ncols + col0] = c00[r] + bv0;
      r0[(size_t)r * Ncols + col1] = c01[r] + bv1;
      r1[(size_t)r * Ncols + col0] = c10[r] + bv0;
      r1[(size_t)r * Ncols + col1] = c11[r] + bv1;
    }
  }
}

// ---------------------------------------------------------------------------
// Fused softmax + trilinear gather.  One wave per (n,q,h); lanes cover the 96
// head channels in 3 strips of 32.  Value volume per (n) is 5.3MB -> L2 hits.
// ---------------------------------------------------------------------------
__global__ void msda_sample(const float* __restrict__ value,  // [ROWS_V][768]
                            const float* __restrict__ offs,   // [ROWS_Q][96]
                            const float* __restrict__ attl,   // [ROWS_Q][32]
                            const float* __restrict__ refp,   // [ROWS_Q][3]
                            float* __restrict__ out_mid) {    // [ROWS_Q][768]
  const int lane = threadIdx.x & 31;
  const size_t wid = (size_t)blockIdx.x * (blockDim.x >> 5) + (threadIdx.x >> 5);
  if (wid >= (size_t)ROWS_Q * HEADS) return;
  const int    h   = (int)(wid & (HEADS - 1));
  const size_t row = wid >> 3;
  const int    n   = (int)(row >> 12);          // row / LQ

  // softmax over P=4 attention logits (every lane redundantly; cheap VALU)
  const float* al = attl + row * 32 + h * 4;
  float l0 = al[0], l1 = al[1], l2 = al[2], l3 = al[3];
  float mx = fmaxf(fmaxf(l0, l1), fmaxf(l2, l3));
  float e0 = __expf(l0 - mx), e1 = __expf(l1 - mx),
        e2 = __expf(l2 - mx), e3 = __expf(l3 - mx);
  float inv = 1.0f / (e0 + e1 + e2 + e3);
  float aw[4] = {e0 * inv, e1 * inv, e2 * inv, e3 * inv};

  const float rx = refp[row * 3 + 0];
  const float ry = refp[row * 3 + 1];
  const float rz = refp[row * 3 + 2];

  const float* vbase = value + ((size_t)n * LIN) * CDIM + h * DHEAD + lane;
  float acc0 = 0.f, acc1 = 0.f, acc2 = 0.f;

#pragma unroll
  for (int p = 0; p < PNTS; ++p) {
    const float* op = offs + row * 96 + (h * PNTS + p) * 3;
    const float inv12 = 1.0f / 12.0f;
    // grid = 2*(ref + off/12) - 1 ; pixel = ((grid+1)*12 - 1)*0.5
    float gx = 2.0f * (rx + op[0] * inv12) - 1.0f;
    float gy = 2.0f * (ry + op[1] * inv12) - 1.0f;
    float gz = 2.0f * (rz + op[2] * inv12) - 1.0f;
    float x = ((gx + 1.0f) * (float)NVOX - 1.0f) * 0.5f;
    float y = ((gy + 1.0f) * (float)NVOX - 1.0f) * 0.5f;
    float z = ((gz + 1.0f) * (float)NVOX - 1.0f) * 0.5f;
    float x0 = floorf(x), y0 = floorf(y), z0 = floorf(z);
#pragma unroll
    for (int c = 0; c < 8; ++c) {
      float xi = x0 + (float)(c & 1);
      float yi = y0 + (float)((c >> 1) & 1);
      float zi = z0 + (float)((c >> 2) & 1);
      float w = (1.0f - fabsf(x - xi)) * (1.0f - fabsf(y - yi)) * (1.0f - fabsf(z - zi));
      bool valid = (xi >= 0.f) & (xi < (float)NVOX) & (yi >= 0.f) &
                   (yi < (float)NVOX) & (zi >= 0.f) & (zi < (float)NVOX);
      float wgt = aw[p] * w * (valid ? 1.0f : 0.0f);
      float xc = fminf(fmaxf(xi, 0.f), (float)(NVOX - 1));
      float yc = fminf(fmaxf(yi, 0.f), (float)(NVOX - 1));
      float zc = fminf(fmaxf(zi, 0.f), (float)(NVOX - 1));
      int idx = (int)((zc * (float)NVOX + yc) * (float)NVOX + xc);
      const float* vp = vbase + (size_t)idx * CDIM;   // clipped => always in range
      acc0 += wgt * vp[0];
      acc1 += wgt * vp[32];
      acc2 += wgt * vp[64];
    }
  }
  float* o = out_mid + row * CDIM + h * DHEAD + lane;
  o[0]  = acc0;
  o[32] = acc1;
  o[64] = acc2;
}

// ---------------------------------------------------------------------------
extern "C" void kernel_launch(void* const* d_in, const int* in_sizes, int n_in,
                              void* d_out, int out_size, void* d_ws, size_t ws_size,
                              hipStream_t stream) {
  const float* query = (const float*)d_in[0];
  const float* refp  = (const float*)d_in[1];
  const float* inpf  = (const float*)d_in[2];
  const float* w_val = (const float*)d_in[3];
  const float* b_val = (const float*)d_in[4];
  const float* w_off = (const float*)d_in[5];
  const float* b_off = (const float*)d_in[6];
  const float* w_att = (const float*)d_in[7];
  const float* b_att = (const float*)d_in[8];
  const float* w_out = (const float*)d_in[9];
  const float* b_out = (const float*)d_in[10];

  // -------- workspace carve-up (256B aligned) --------
  char* ws = (char*)d_ws;
  size_t off = 0;
  auto carve = [&](size_t bytes) -> char* {
    char* p = ws + off;
    off = (off + bytes + 255) & ~(size_t)255;
    return p;
  };
  const size_t mtQ = ROWS_Q / 16, mtV = ROWS_V / 16;
  _Float16* pA_q   = (_Float16*)carve(mtQ * KTILES * 512 * sizeof(_Float16)); // 50.3MB (reused for out_mid)
  _Float16* pA_v   = (_Float16*)carve(mtV * KTILES * 512 * sizeof(_Float16)); // 21.2MB
  float*    valueF = (float*)   carve((size_t)ROWS_V * CDIM * sizeof(float)); // 42.5MB
  float*    offsF  = (float*)   carve((size_t)ROWS_Q * 96 * sizeof(float));   // 12.6MB
  float*    attlF  = (float*)   carve((size_t)ROWS_Q * 32 * sizeof(float));   //  4.2MB
  float*    outMid = (float*)   carve((size_t)ROWS_Q * CDIM * sizeof(float)); // 100.7MB
  _Float16* pB_val = (_Float16*)carve((size_t)48 * KTILES * 512 * sizeof(_Float16));
  _Float16* pB_off = (_Float16*)carve((size_t)6  * KTILES * 512 * sizeof(_Float16));
  _Float16* pB_att = (_Float16*)carve((size_t)2  * KTILES * 512 * sizeof(_Float16));
  _Float16* pB_out = (_Float16*)carve((size_t)48 * KTILES * 512 * sizeof(_Float16));
  (void)ws_size; (void)in_sizes; (void)n_in; (void)out_size;

  const dim3 blk(256);
  auto blocks = [](size_t total) { return dim3((unsigned)((total + 255) / 256)); };

  // -------- pack weights (B operands) --------
  pack_b_f16<<<blocks((size_t)48 * KTILES * 512), blk, 0, stream>>>(w_val, pB_val, 768, (size_t)48 * KTILES * 512);
  pack_b_f16<<<blocks((size_t)6  * KTILES * 512), blk, 0, stream>>>(w_off, pB_off, 96,  (size_t)6  * KTILES * 512);
  pack_b_f16<<<blocks((size_t)2  * KTILES * 512), blk, 0, stream>>>(w_att, pB_att, 32,  (size_t)2  * KTILES * 512);
  pack_b_f16<<<blocks((size_t)48 * KTILES * 512), blk, 0, stream>>>(w_out, pB_out, 768, (size_t)48 * KTILES * 512);

  // -------- value = input_flatten @ w_val + b_val --------
  pack_a_f16<<<blocks(mtV * KTILES * 512), blk, 0, stream>>>(inpf, pA_v, CDIM, mtV * KTILES * 512);
  // Mtiles=864 Ntiles=48: waves 2x4 -> WG covers 64 rows x 128 cols
  wmma_gemm_f16<<<dim3(48 / 8, 864 / 4), blk, 0, stream>>>(pA_v, pB_val, b_val, valueF, 768, 4);

  // -------- offsets & attention logits from query --------
  pack_a_f16<<<blocks(mtQ * KTILES * 512), blk, 0, stream>>>(query, pA_q, CDIM, mtQ * KTILES * 512);
  // Mtiles=2048 Ntiles=6: waves 8x1 -> WG covers 256 rows x 32 cols
  wmma_gemm_f16<<<dim3(6 / 2, 2048 / 16), blk, 0, stream>>>(pA_q, pB_off, b_off, offsF, 96, 1);
  // Ntiles=2
  wmma_gemm_f16<<<dim3(2 / 2, 2048 / 16), blk, 0, stream>>>(pA_q, pB_att, b_att, attlF, 32, 1);

  // -------- softmax + trilinear sampling --------
  msda_sample<<<dim3((ROWS_Q * HEADS) / 8), blk, 0, stream>>>(valueF, offsF, attlF, refp, outMid);

  // -------- final projection: out = out_mid @ w_out + b_out --------
  pack_a_f16<<<blocks(mtQ * KTILES * 512), blk, 0, stream>>>(outMid, pA_q, CDIM, mtQ * KTILES * 512);
  wmma_gemm_f16<<<dim3(48 / 8, 2048 / 4), blk, 0, stream>>>(pA_q, pB_out, b_out, (float*)d_out, 768, 4);
}
